// GLBLMLPLayer_60679297957929
// MI455X (gfx1250) — compile-verified, matrix operand order checked by reading, not run on hardware
//
#include <hip/hip_runtime.h>
#include <math.h>

// problem constants
#define NTOK 4096   // B*S
#define DM   512
#define FM   2048
#define EM   16

typedef _Float16 v16h __attribute__((ext_vector_type(16)));
typedef _Float16 h8   __attribute__((ext_vector_type(8)));
typedef float    v8f  __attribute__((ext_vector_type(8)));

// Load a 16x32 (f16) WMMA A/B fragment whose K dimension is contiguous at p.
// Per ISA layout: lane<16 holds K {0..7, 16..23}, lane>=16 holds K {8..15, 24..31};
// caller adds the per-lane +8 offset (khi) into p.
__device__ __forceinline__ v16h ldfrag(const _Float16* p) {
  h8 lo = *(const h8*)p;         // K base+0..7   (16B aligned)
  h8 hi = *(const h8*)(p + 16);  // K base+16..23
  return __builtin_shufflevector(lo, hi, 0,1,2,3,4,5,6,7,8,9,10,11,12,13,14,15);
}

__device__ __forceinline__ v8f wmma16(v16h a, v16h b, v8f c) {
  return __builtin_amdgcn_wmma_f32_16x16x32_f16(false, a, false, b, (short)0, c, false, false);
}

__device__ __forceinline__ float gelu_exact(float v) {
  return 0.5f * v * (1.0f + erff(v * 0.70710678118654752f));
}

// ---------------- weight prep: transpose + f32->f16 ----------------
// dst[e][f][d] = up_w[e][d][f]
__global__ __launch_bounds__(256) void k_prep_up(const float* __restrict__ up_w,
                                                 _Float16* __restrict__ dst) {
  unsigned i = blockIdx.x * 256u + threadIdx.x;   // E*F*D = 16,777,216
  unsigned d = i & (DM - 1);
  unsigned f = (i >> 9) & (FM - 1);
  unsigned e = i >> 20;
  dst[i] = (_Float16)up_w[((size_t)e * DM + d) * FM + f];
}

// dst[e][d][f] = down_w[e][f][d]
__global__ __launch_bounds__(256) void k_prep_down(const float* __restrict__ down_w,
                                                   _Float16* __restrict__ dst) {
  unsigned i = blockIdx.x * 256u + threadIdx.x;
  unsigned f = i & (FM - 1);
  unsigned d = (i >> 11) & (DM - 1);
  unsigned e = i >> 20;
  dst[i] = (_Float16)down_w[((size_t)e * FM + f) * DM + d];
}

// ---------------- router: gate MLP + softmax + top-2 ----------------
__global__ __launch_bounds__(256) void k_router(
    const float* __restrict__ x, const float* __restrict__ gw1,
    const float* __restrict__ gb1, const float* __restrict__ gw2,
    const float* __restrict__ gb2, float* __restrict__ rp_out,
    float* __restrict__ pcomb, float* __restrict__ usage) {
  __shared__ float xs[DM];
  __shared__ float hs[DM / 2];
  __shared__ float sc[EM];
  const int token = blockIdx.x;
  const int t = threadIdx.x;

  xs[t]       = x[(size_t)token * DM + t];
  xs[t + 256] = x[(size_t)token * DM + 256 + t];
  __syncthreads();

  float s = gb1[t];
  for (int d = 0; d < DM; ++d) s += xs[d] * gw1[d * 256 + t];
  hs[t] = fmaxf(s, 0.0f);
  __syncthreads();

  if (t < EM) {
    float sv = gb2[t];
    for (int j = 0; j < 256; ++j) sv += hs[j] * gw2[j * EM + t];
    sc[t] = sv;
  }
  __syncthreads();

  if (t == 0) {
    float pr[EM];
    float mx = sc[0];
    for (int e = 1; e < EM; ++e) mx = fmaxf(mx, sc[e]);
    float sum = 0.0f;
    for (int e = 0; e < EM; ++e) { pr[e] = expf((sc[e] - mx) * 2.0f); sum += pr[e]; } // /T, T=0.5
    float inv = 1.0f / sum;
    for (int e = 0; e < EM; ++e) {
      pr[e] *= inv;
      rp_out[(size_t)token * EM + e] = pr[e];
      atomicAdd(&usage[e], pr[e]);
    }
    // top-2 (first-occurrence on ties, matching lax.top_k)
    int i0 = 0;
    for (int e = 1; e < EM; ++e) if (pr[e] > pr[i0]) i0 = e;
    int i1 = -1;
    for (int e = 0; e < EM; ++e) {
      if (e == i0) continue;
      if (i1 < 0 || pr[e] > pr[i1]) i1 = e;
    }
    // softmax over the two selected probabilities
    float b = expf(pr[i1] - pr[i0]);
    float p0 = 1.0f / (1.0f + b);
    float* pc = pcomb + (size_t)token * EM;
    for (int e = 0; e < EM; ++e) pc[e] = 0.0f;
    pc[i0] = p0;
    pc[i1] = 1.0f - p0;
  }
}

__global__ void k_loss(const float* __restrict__ usage, float* __restrict__ out_loss) {
  if (threadIdx.x == 0 && blockIdx.x == 0) {
    float acc = 0.0f;
    for (int e = 0; e < EM; ++e) {
      float u = usage[e] * (1.0f / (float)NTOK) - (1.0f / (float)EM);
      acc += u * u;
    }
    *out_loss = acc * (1.0f / (float)EM);
  }
}

// ---------------- fused dense-MoE expert kernel ----------------
// 128 blocks, each owns 32 tokens x 512 out dims; 8 waves, wave w owns
// D columns [w*64, w*64+64). Loops all 16 experts; hidden rows are pre-scaled
// by the combine weight so one fp32 accumulator survives all experts.
__global__ __launch_bounds__(256) void k_moe(
    const float* __restrict__ x,        // [T][D] f32
    const _Float16* __restrict__ WUt,   // [E][F][D] f16 (transposed up_w)
    const _Float16* __restrict__ WDt,   // [E][D][F] f16 (transposed down_w)
    const float* __restrict__ up_b,     // [E][F]
    const float* __restrict__ down_b,   // [E][D]
    const float* __restrict__ pcomb,    // [T][E]
    float* __restrict__ out) {          // [T][D]
  __shared__ __align__(16) _Float16 XS[32 * DM];   // 32 KB: x tile in f16
  __shared__ __align__(16) _Float16 HT[32 * 128];  // 8 KB: scaled hidden chunk
  __shared__ float PV[32 * EM];                    // combine weights
  __shared__ float PB[32 * DM];                    // sum_e p*down_b

  const int tid  = threadIdx.x;
  const int tile = blockIdx.x;          // token tile of 32
  const int lane = tid & 31;
  const int wv   = tid >> 5;            // wave 0..7
  const int ll   = lane & 15;
  const int khi  = (lane >> 4) << 3;    // per-lane K offset (0 or 8), == 8*laneHi

  // combine weights for this tile (32 tokens x 16 experts = 512 contiguous floats)
  PV[tid]       = pcomb[(size_t)tile * 512 + tid];
  PV[tid + 256] = pcomb[(size_t)tile * 512 + 256 + tid];

  // x tile -> f16 LDS
  for (int i = tid; i < 32 * DM; i += 256)
    XS[i] = (_Float16)x[(size_t)tile * 32 * DM + i];
  __syncthreads();

  // PB[m][d] = sum_e PV[m][e] * down_b[e][d]
  for (int rep = 0; rep < 2; ++rep) {
    int d = tid + rep * 256;
    float db[EM];
#pragma unroll
    for (int e = 0; e < EM; ++e) db[e] = down_b[e * DM + d];
    for (int m = 0; m < 32; ++m) {
      float s = 0.0f;
#pragma unroll
      for (int e = 0; e < EM; ++e) s += PV[m * EM + e] * db[e];
      PB[m * DM + d] = s;
    }
  }

  v8f acc[2][4];
#pragma unroll
  for (int i = 0; i < 2; ++i)
#pragma unroll
    for (int j = 0; j < 4; ++j) acc[i][j] = (v8f){0, 0, 0, 0, 0, 0, 0, 0};

  for (int e = 0; e < EM; ++e) {
    const _Float16* WU = WUt + (size_t)e * FM * DM;
    const _Float16* WD = WDt + (size_t)e * DM * FM;
    for (int fc = 0; fc < FM; fc += 128) {
      // ---- stage 1: H[32,128] = gelu(X @ up_w + b) * p ----
      v8f h0 = (v8f){0, 0, 0, 0, 0, 0, 0, 0};
      v8f h1 = (v8f){0, 0, 0, 0, 0, 0, 0, 0};
      const int col = wv * 16 + ll;                         // chunk-local F column
      const _Float16* bb  = WU + (size_t)(fc + col) * DM + khi;
      const _Float16* a0p = XS + ll * DM + khi;
      const _Float16* a1p = XS + (16 + ll) * DM + khi;
#pragma unroll 4
      for (int kk = 0; kk < DM; kk += 32) {
        v16h bf = ldfrag(bb + kk);
        v16h a0 = ldfrag(a0p + kk);
        v16h a1 = ldfrag(a1p + kk);
        h0 = wmma16(a0, bf, h0);
        h1 = wmma16(a1, bf, h1);
      }
      float bias = up_b[e * FM + fc + col];
#pragma unroll
      for (int r = 0; r < 8; ++r) {
        int m0 = r + khi;                 // C layout: M = r + 8*laneHi, N = lane&15
        float v0 = gelu_exact(h0[r] + bias);
        HT[m0 * 128 + col] = (_Float16)(v0 * PV[m0 * EM + e]);
        int m1 = 16 + m0;
        float v1 = gelu_exact(h1[r] + bias);
        HT[m1 * 128 + col] = (_Float16)(v1 * PV[m1 * EM + e]);
      }
      __syncthreads();
      // ---- stage 2: acc += H @ down_w (wave's 64 D-cols) ----
      const _Float16* ha0 = HT + ll * 128 + khi;
      const _Float16* ha1 = HT + (16 + ll) * 128 + khi;
#pragma unroll
      for (int kk = 0; kk < 128; kk += 32) {
        v16h a0 = ldfrag(ha0 + kk);
        v16h a1 = ldfrag(ha1 + kk);
#pragma unroll
        for (int nt = 0; nt < 4; ++nt) {
          int d = wv * 64 + nt * 16 + ll;
          v16h bf = ldfrag(WD + (size_t)d * FM + fc + kk + khi);
          acc[0][nt] = wmma16(a0, bf, acc[0][nt]);
          acc[1][nt] = wmma16(a1, bf, acc[1][nt]);
        }
      }
      __syncthreads();
    }
  }

  // epilogue: add bias-combine term and store
#pragma unroll
  for (int mt = 0; mt < 2; ++mt)
#pragma unroll
    for (int nt = 0; nt < 4; ++nt)
#pragma unroll
      for (int r = 0; r < 8; ++r) {
        int m = mt * 16 + r + khi;
        int d = wv * 64 + nt * 16 + ll;
        out[((size_t)tile * 32 + m) * DM + d] = acc[mt][nt][r] + PB[m * DM + d];
      }
}

extern "C" void kernel_launch(void* const* d_in, const int* in_sizes, int n_in,
                              void* d_out, int out_size, void* d_ws, size_t ws_size,
                              hipStream_t stream) {
  (void)in_sizes; (void)n_in; (void)out_size; (void)ws_size;
  const float* x      = (const float*)d_in[0];
  const float* gw1    = (const float*)d_in[1];
  const float* gb1    = (const float*)d_in[2];
  const float* gw2    = (const float*)d_in[3];
  const float* gb2    = (const float*)d_in[4];
  const float* up_w   = (const float*)d_in[5];
  const float* up_b   = (const float*)d_in[6];
  const float* down_w = (const float*)d_in[7];
  const float* down_b = (const float*)d_in[8];

  float* out = (float*)d_out;
  float* out_main = out;                       // [4096][512]
  float* out_loss = out + (size_t)NTOK * DM;   // scalar
  float* rp_out   = out_loss + 1;              // [4096][16]

  char* ws = (char*)d_ws;
  _Float16* wu    = (_Float16*)ws;                                   // 32 MB
  _Float16* wd    = (_Float16*)(ws + (size_t)EM * FM * DM * 2);      // 32 MB
  float*    pcomb = (float*)(ws + (size_t)EM * FM * DM * 4);         // 256 KB
  float*    usage = (float*)(ws + (size_t)EM * FM * DM * 4 + (size_t)NTOK * EM * 4);

  hipMemsetAsync(usage, 0, EM * sizeof(float), stream);

  const int prep_blocks = (EM * FM * DM) / 256;  // 65536
  k_prep_up  <<<prep_blocks, 256, 0, stream>>>(up_w, wu);
  k_prep_down<<<prep_blocks, 256, 0, stream>>>(down_w, wd);

  k_router<<<NTOK, 256, 0, stream>>>(x, gw1, gb1, gw2, gb2, rp_out, pcomb, usage);
  k_loss<<<1, 32, 0, stream>>>(usage, out_loss);

  k_moe<<<NTOK / 32, 256, 0, stream>>>(x, wu, wd, up_b, down_b, pcomb, out_main);
}
